// DMatrix_31868657336793
// MI455X (gfx1250) — compile-verified, matrix operand order checked by reading, not run on hardware
//
#include <hip/hip_runtime.h>
#include <hip/hip_bf16.h>

// ---------------------------------------------------------------------------
// Pairwise squared Euclidean distance on MI455X (gfx1250):
//   D[b,n,m] = ||A[b,n]||^2 - 2 * <A[b,n], B[b,m]> + ||B[b,m]||^2
// Shapes: B=16, N=4096, K=256, fp32 in / fp32 out.
// Strategy: bf16x2 split GEMM on v_wmma_f32_16x16x32_bf16 (3 WMMA per tile:
// hi*hi + hi*lo + lo*hi, fp32 accumulate) + fused norm epilogue.
// ---------------------------------------------------------------------------

typedef __attribute__((ext_vector_type(16))) __bf16        v16bf;
typedef __attribute__((ext_vector_type(8)))  float         v8f;
typedef __attribute__((ext_vector_type(8)))  unsigned short us8;

union FragU { v16bf bf; us8 u[2]; };

#define BATCH 16
#define NDIM  4096
#define KDIM  256
#define BM    128
#define BN    128
#define BK    32
#define LROW  40               // shorts per LDS row: 32 data + 8 pad (80 B)
#define PART  (BM * LROW)      // 5120 shorts per tile-part

static __device__ __forceinline__ unsigned short f32_to_bf16_rne(float x) {
    unsigned int u = __float_as_uint(x);
    unsigned int r = u + 0x7FFFu + ((u >> 16) & 1u);
    return (unsigned short)(r >> 16);
}
static __device__ __forceinline__ float bf16_bits_to_f32(unsigned short b) {
    return __uint_as_float(((unsigned int)b) << 16);
}

// Split one float4 into bf16 hi/lo and store packed (ds_store_b64 each).
static __device__ __forceinline__ void split_store(unsigned short* ldsHi,
                                                   unsigned short* ldsLo,
                                                   int row, int c4, float4 v) {
    float f[4] = {v.x, v.y, v.z, v.w};
    unsigned short h[4], l[4];
#pragma unroll
    for (int j = 0; j < 4; ++j) {
        unsigned short hb = f32_to_bf16_rne(f[j]);
        h[j] = hb;
        l[j] = f32_to_bf16_rne(f[j] - bf16_bits_to_f32(hb));
    }
    unsigned int hp0 = (unsigned int)h[0] | ((unsigned int)h[1] << 16);
    unsigned int hp1 = (unsigned int)h[2] | ((unsigned int)h[3] << 16);
    unsigned int lp0 = (unsigned int)l[0] | ((unsigned int)l[1] << 16);
    unsigned int lp1 = (unsigned int)l[2] | ((unsigned int)l[3] << 16);
    uint2* ph = (uint2*)(ldsHi + row * LROW + c4 * 4);   // 8B aligned (80|8)
    uint2* pl = (uint2*)(ldsLo + row * LROW + c4 * 4);
    *ph = make_uint2(hp0, hp1);
    *pl = make_uint2(lp0, lp1);
}

// ISA 16-bit A-matrix 16x32 fragment layout (also used for B since B is
// stored row-major [n,k] and we compute A * B^T):
//   lane M = lane&15; lanes<16 hold K 0..7 & 16..23; lanes>=16 hold 8..15 & 24..31
//   -> two contiguous 16-byte chunks at K = 8*hf and 16 + 8*hf.
static __device__ __forceinline__ v16bf load_frag(const unsigned short* part,
                                                  int row, int hf) {
    FragU f;
    f.u[0] = *(const us8*)(part + row * LROW + 8 * hf);        // 16B aligned
    f.u[1] = *(const us8*)(part + row * LROW + 16 + 8 * hf);
    return f.bf;
}

// ---------------------------------------------------------------------------
// Row-norm kernel: one wave32 per row of 256 floats. 8 rows per block.
// ---------------------------------------------------------------------------
__global__ __launch_bounds__(256) void row_norms_kernel(
    const float* __restrict__ X, float* __restrict__ r, int nrows) {
    int wid  = threadIdx.x >> 5;
    int lane = threadIdx.x & 31;
    int row  = blockIdx.x * 8 + wid;
    if (row >= nrows) return;
    const float4* p = (const float4*)(X + (size_t)row * KDIM);
    float s = 0.f;
#pragma unroll
    for (int i = 0; i < 2; ++i) {
        float4 v = p[lane + 32 * i];
        s += v.x * v.x + v.y * v.y + v.z * v.z + v.w * v.w;
    }
#pragma unroll
    for (int off = 16; off > 0; off >>= 1) s += __shfl_xor(s, off, 32);
    if (lane == 0) r[row] = s;
}

// ---------------------------------------------------------------------------
// Main fused distance-GEMM kernel. Block = 256 threads (8 waves), 128x128 tile.
// ---------------------------------------------------------------------------
__global__ __launch_bounds__(256) void dist_gemm_kernel(
    const float* __restrict__ A, const float* __restrict__ B,
    const float* __restrict__ rA, const float* __restrict__ rB,
    float* __restrict__ out) {
    // LDS parts: 0=A_hi 1=A_lo 2=B_hi 3=B_lo (40 KB total, padded rows)
    __shared__ unsigned short lds[4 * PART];

    const int bz   = blockIdx.z;
    const int m0   = blockIdx.y * BM;
    const int n0   = blockIdx.x * BN;
    const int tid  = threadIdx.x;
    const int wave = tid >> 5;
    const int lane = tid & 31;
    const int hf   = lane >> 4;
    const int lrow = lane & 15;
    const int wm   = wave >> 2;  // 0..1 -> 64-row slab
    const int wn   = wave & 3;   // 0..3 -> 32-col slab

    const float* Abase = A + ((size_t)bz * NDIM + m0) * KDIM;
    const float* Bbase = B + ((size_t)bz * NDIM + n0) * KDIM;

    v8f acc[4][2];
#pragma unroll
    for (int mt = 0; mt < 4; ++mt)
#pragma unroll
        for (int nt = 0; nt < 2; ++nt)
            acc[mt][nt] = v8f{0.f, 0.f, 0.f, 0.f, 0.f, 0.f, 0.f, 0.f};

    for (int k0 = 0; k0 < KDIM; k0 += BK) {
        // ---- stage fp32 panels -> bf16 hi/lo in LDS (coalesced float4) ----
#pragma unroll
        for (int it = 0; it < 4; ++it) {
            int fid = tid + 256 * it;     // float4 id 0..1023
            int row = fid >> 3;           // 128 rows
            int c4  = fid & 7;            // 8 float4 per 32-wide row
            float4 va = *(const float4*)(Abase + (size_t)row * KDIM + k0 + c4 * 4);
            float4 vb = *(const float4*)(Bbase + (size_t)row * KDIM + k0 + c4 * 4);
            split_store(lds + 0 * PART, lds + 1 * PART, row, c4, va);
            split_store(lds + 2 * PART, lds + 3 * PART, row, c4, vb);
        }
        __syncthreads();

        // ---- fragments + WMMA ----
        v16bf afh[4], afl[4], bfh[2], bfl[2];
#pragma unroll
        for (int mt = 0; mt < 4; ++mt) {
            int r = wm * 64 + mt * 16 + lrow;
            afh[mt] = load_frag(lds + 0 * PART, r, hf);
            afl[mt] = load_frag(lds + 1 * PART, r, hf);
        }
#pragma unroll
        for (int nt = 0; nt < 2; ++nt) {
            int r = wn * 32 + nt * 16 + lrow;
            bfh[nt] = load_frag(lds + 2 * PART, r, hf);
            bfl[nt] = load_frag(lds + 3 * PART, r, hf);
        }
#pragma unroll
        for (int mt = 0; mt < 4; ++mt)
#pragma unroll
            for (int nt = 0; nt < 2; ++nt) {
                acc[mt][nt] = __builtin_amdgcn_wmma_f32_16x16x32_bf16(
                    false, afh[mt], false, bfh[nt], (short)0, acc[mt][nt], false, false);
                acc[mt][nt] = __builtin_amdgcn_wmma_f32_16x16x32_bf16(
                    false, afh[mt], false, bfl[nt], (short)0, acc[mt][nt], false, false);
                acc[mt][nt] = __builtin_amdgcn_wmma_f32_16x16x32_bf16(
                    false, afl[mt], false, bfh[nt], (short)0, acc[mt][nt], false, false);
            }
        __syncthreads();
    }

    // ---- epilogue: D = rA + rB - 2*dot  (C/D layout: M = 8*hf + i, N = lrow)
    const float* rAp = rA + (size_t)bz * NDIM;
    const float* rBp = rB + (size_t)bz * NDIM;
#pragma unroll
    for (int mt = 0; mt < 4; ++mt) {
        int gmBase = m0 + wm * 64 + mt * 16 + 8 * hf;
#pragma unroll
        for (int nt = 0; nt < 2; ++nt) {
            int   gn = n0 + wn * 32 + nt * 16 + lrow;
            float rb = rBp[gn];
#pragma unroll
            for (int i = 0; i < 8; ++i) {
                int gm = gmBase + i;
                float v = rAp[gm] + rb - 2.0f * acc[mt][nt][i];
                out[((size_t)bz * NDIM + gm) * NDIM + gn] = v;
            }
        }
    }
}

// ---------------------------------------------------------------------------
extern "C" void kernel_launch(void* const* d_in, const int* in_sizes, int n_in,
                              void* d_out, int out_size, void* d_ws, size_t ws_size,
                              hipStream_t stream) {
    const float* A = (const float*)d_in[0];
    const float* B = (const float*)d_in[1];
    float* out = (float*)d_out;

    // workspace: rA (16*4096 f32) then rB (16*4096 f32) = 512 KB
    float* rA = (float*)d_ws;
    float* rB = rA + BATCH * NDIM;

    const int nrows = BATCH * NDIM;                 // 65536
    row_norms_kernel<<<nrows / 8, 256, 0, stream>>>(A, rA, nrows);
    row_norms_kernel<<<nrows / 8, 256, 0, stream>>>(B, rB, nrows);

    dim3 grid(NDIM / BN, NDIM / BM, BATCH);         // 32 x 32 x 16
    dist_gemm_kernel<<<grid, 256, 0, stream>>>(A, B, rA, rB, out);
}
